// FTE_42949673220
// MI455X (gfx1250) — matrix operand order repaired and checked by reading, not compiled, not run on hardware
//
#include <hip/hip_runtime.h>
#include <hip/hip_bf16.h>
#include <math.h>

// H = 256 throughout.
#define HDIM 256

typedef __bf16 bf16_t;
typedef __attribute__((ext_vector_type(16))) __bf16 v16bf;
typedef __attribute__((ext_vector_type(8)))  __bf16 v8bf;
typedef __attribute__((ext_vector_type(8)))  float  v8f;

union AFrag { v16bf v; v8bf h[2]; };

// A-matrix (16x32 bf16) fragment from a row-major bf16 buffer (row stride ld elements).
// lanes 0-15 row M=lane, K = 0..15; lanes 16-31 row M=lane-16, K = 16..31 (matches B fragment).
__device__ __forceinline__ v16bf load_a_frag(const bf16_t* base, int ld, int k0, int lane) {
    const int m  = lane & 15;
    const int kb = (lane >> 4) << 4;            // 0 or 16
    const bf16_t* p = base + m * ld + k0 + kb;
    AFrag a;
    a.h[0] = *(const v8bf*)(p);                 // K = k0+kb   .. k0+kb+7
    a.h[1] = *(const v8bf*)(p + 8);             // K = k0+kb+8 .. k0+kb+15
    return a.v;
}

// B-matrix (32x16 bf16) fragment from transposed (N-major) weights Wt[N][Kdim].
// lanes 0-15: column n0+lane, K = k0..k0+15; lanes 16-31: column n0+lane-16, K = k0+16..k0+31.
__device__ __forceinline__ v16bf load_b_frag(const bf16_t* Wt, int Kdim, int n0, int k0, int lane) {
    const int n  = n0 + (lane & 15);
    const int ko = k0 + ((lane >> 4) << 4);     // 0 or 16
    return *(const v16bf*)(Wt + (size_t)n * Kdim + ko);
}

__device__ __forceinline__ v8f wmma_bf16(v16bf a, v16bf b, v8f c) {
    return __builtin_amdgcn_wmma_f32_16x16x32_bf16(false, a, false, b, (short)0, c, false, false);
}

// ---------------------------------------------------------------------------
// Prep: transpose weights into bf16 N-major layout in workspace.
//   WtE[512][256] = W_equi^T ; Wt1[256][512] = W1^T ; Wt2[768][256] = W2^T
// ---------------------------------------------------------------------------
__global__ void prep_weights(const float* __restrict__ WE, const float* __restrict__ W1,
                             const float* __restrict__ W2,
                             bf16_t* __restrict__ WtE, bf16_t* __restrict__ Wt1,
                             bf16_t* __restrict__ Wt2) {
    int idx = blockIdx.x * 256 + threadIdx.x;
    if (idx < 512 * 256) {                       // W_equi (256 x 512) -> (512 x 256)
        int n = idx >> 8, k = idx & 255;
        WtE[idx] = (bf16_t)WE[k * 512 + n];
        return;
    }
    idx -= 512 * 256;
    if (idx < 256 * 512) {                       // W1 (512 x 256) -> (256 x 512)
        int n = idx >> 9, k = idx & 511;
        Wt1[idx] = (bf16_t)W1[k * 256 + n];
        return;
    }
    idx -= 256 * 512;
    if (idx < 768 * 256) {                       // W2 (256 x 768) -> (768 x 256)
        int n = idx >> 8, k = idx & 255;
        Wt2[idx] = (bf16_t)W2[k * 768 + n];
        return;
    }
}

// ---------------------------------------------------------------------------
// Fused kernel: one block = 16 rows of n, 128 threads = 4 waves (wave32).
// LDS layout (dynamic, 90112 B):
//   A1  bf16 [3*16][256]  staged vec tiles          (24576 B)
//   Hb  bf16 [16][512]    h = [x | scalar]          (16384 B)
//   H2  bf16 [16][256]    silu activation           ( 8192 B)
//   V2  bf16 [3*16][256]  vec2                      (24576 B)
//   VD  f32  [16][256]    vec_dot                   (16384 B)
// ---------------------------------------------------------------------------
__global__ void __launch_bounds__(128) fused_gated_update(
    const float* __restrict__ xg,        // (N, 256)
    const float* __restrict__ vecg,      // (N, 3, 256)
    const float* __restrict__ b1g,       // (256)
    const float* __restrict__ b2g,       // (768)
    const bf16_t* __restrict__ WtE,      // (512, 256)
    const bf16_t* __restrict__ Wt1,      // (256, 512)
    const bf16_t* __restrict__ Wt2,      // (768, 256)
    float* __restrict__ outp,            // dx (N,256) ++ dvec (N,3,256)
    int N)
{
    extern __shared__ __align__(16) char smem_raw[];
    bf16_t* A1 = (bf16_t*)smem_raw;                 // 3*16*256
    bf16_t* Hb = A1 + 3 * 16 * 256;                 // 16*512
    bf16_t* H2 = Hb + 16 * 512;                     // 16*256
    bf16_t* V2 = H2 + 16 * 256;                     // 3*16*256
    float*  VD = (float*)(V2 + 3 * 16 * 256);       // 16*256

    const int tid  = threadIdx.x;
    const int lane = tid & 31;
    const int wid  = tid >> 5;
    const int n0   = blockIdx.x * 16;

    // L2 prefetch of the weight panels this block will stream.
    __builtin_prefetch(WtE + (size_t)tid * 512, 0, 1);
    __builtin_prefetch(Wt2 + (size_t)tid * 512, 0, 1);

    // ---- Phase 1: stage x -> Hb[:, 0:256], vec -> A1 (f32 -> bf16) -------------
    for (int i = tid; i < 16 * 64; i += 128) {       // 1024 float4 of x
        int m  = i >> 6;
        int c4 = (i & 63) << 2;
        float4 v = *(const float4*)(xg + (size_t)(n0 + m) * HDIM + c4);
        bf16_t* d = Hb + m * 512 + c4;
        d[0] = (bf16_t)v.x; d[1] = (bf16_t)v.y; d[2] = (bf16_t)v.z; d[3] = (bf16_t)v.w;
    }
    for (int i = tid; i < 48 * 64; i += 128) {       // 3072 float4 of vec
        int r  = i >> 6;                             // r = c*16 + m
        int c4 = (i & 63) << 2;
        int c = r >> 4, m = r & 15;
        float4 v = *(const float4*)(vecg + ((size_t)(n0 + m) * 3 + c) * HDIM + c4);
        bf16_t* d = A1 + r * HDIM + c4;
        d[0] = (bf16_t)v.x; d[1] = (bf16_t)v.y; d[2] = (bf16_t)v.z; d[3] = (bf16_t)v.w;
    }
    __syncthreads();

    const int col = (lane & 15);
    const int mb  = (lane >> 4) << 3;                // C/D frag: M = mb + j

    // ---- Phase 2: GEMM1 (vec @ W_equi) + scalar/vec_dot/vec2 ------------------
    const float inv_sqrt_h = 0.0625f;                // 1/sqrt(256)
    for (int t = wid; t < 16; t += 4) {
        v8f acc1[3] = {}, acc2[3] = {};
        #pragma unroll
        for (int kk = 0; kk < 8; ++kk) {
            const int k0 = kk * 32;
            v16bf bA = load_b_frag(WtE, 256, t * 16, k0, lane);        // vec1 cols
            v16bf bB = load_b_frag(WtE, 256, 256 + t * 16, k0, lane);  // vec2 cols
            #pragma unroll
            for (int c = 0; c < 3; ++c) {
                v16bf af = load_a_frag(A1 + c * 16 * HDIM, HDIM, k0, lane);
                acc1[c] = wmma_bf16(af, bA, acc1[c]);
                acc2[c] = wmma_bf16(af, bB, acc2[c]);
            }
        }
        const int cc = t * 16 + col;
        #pragma unroll
        for (int j = 0; j < 8; ++j) {
            const int m = mb + j;
            float s0 = acc1[0][j], s1 = acc1[1][j], s2 = acc1[2][j];
            float u0 = acc2[0][j], u1 = acc2[1][j], u2 = acc2[2][j];
            Hb[m * 512 + 256 + cc] = (bf16_t)sqrtf(s0 * s0 + s1 * s1 + s2 * s2);
            VD[m * 256 + cc] = (s0 * u0 + s1 * u1 + s2 * u2) * inv_sqrt_h;
            V2[(0 * 16 + m) * 256 + cc] = (bf16_t)u0;
            V2[(1 * 16 + m) * 256 + cc] = (bf16_t)u1;
            V2[(2 * 16 + m) * 256 + cc] = (bf16_t)u2;
        }
    }
    __syncthreads();

    // ---- Phase 3: GEMM2 ([x|scalar] @ W1) + bias + silu -----------------------
    for (int nt = wid; nt < 16; nt += 4) {
        v8f acc = {};
        #pragma unroll
        for (int kk = 0; kk < 16; ++kk) {
            v16bf af = load_a_frag(Hb, 512, kk * 32, lane);
            v16bf bf = load_b_frag(Wt1, 512, nt * 16, kk * 32, lane);
            acc = wmma_bf16(af, bf, acc);
        }
        const int cc = nt * 16 + col;
        const float bias = b1g[cc];
        #pragma unroll
        for (int j = 0; j < 8; ++j) {
            float v = acc[j] + bias;
            H2[(mb + j) * 256 + cc] = (bf16_t)(v / (1.0f + __expf(-v)));
        }
    }
    __syncthreads();

    // ---- Phase 4: GEMM3 (h @ W2) + epilogue (dx, dvec) ------------------------
    const float inv_sqrt_2 = 0.70710678118654752f;
    const size_t dvec_base = (size_t)N * 256;
    for (int t = wid; t < 16; t += 4) {
        v8f a1 = {}, a2 = {}, a3 = {};
        #pragma unroll
        for (int kk = 0; kk < 8; ++kk) {
            const int k0 = kk * 32;
            v16bf af = load_a_frag(H2, 256, k0, lane);
            a1 = wmma_bf16(af, load_b_frag(Wt2, 256, t * 16, k0, lane), a1);
            a2 = wmma_bf16(af, load_b_frag(Wt2, 256, 256 + t * 16, k0, lane), a2);
            a3 = wmma_bf16(af, load_b_frag(Wt2, 256, 512 + t * 16, k0, lane), a3);
        }
        const int cc = t * 16 + col;
        const float bb1 = b2g[cc], bb2 = b2g[cc + 256], bb3 = b2g[cc + 512];
        #pragma unroll
        for (int j = 0; j < 8; ++j) {
            const int m = mb + j;
            const size_t row = (size_t)(n0 + m);
            float dx = (a1[j] + bb1 + a2[j] + bb2 + VD[m * 256 + cc]) * inv_sqrt_2;
            outp[row * 256 + cc] = dx;
            float x3 = a3[j] + bb3;
            #pragma unroll
            for (int c = 0; c < 3; ++c) {
                float v2v = (float)V2[(c * 16 + m) * 256 + cc];
                outp[dvec_base + (row * 3 + c) * 256 + cc] = x3 * v2v;
            }
        }
    }
}

// ---------------------------------------------------------------------------
extern "C" void kernel_launch(void* const* d_in, const int* in_sizes, int n_in,
                              void* d_out, int out_size, void* d_ws, size_t ws_size,
                              hipStream_t stream) {
    const float* xg   = (const float*)d_in[0];   // (N, 256)
    const float* vecg = (const float*)d_in[1];   // (N, 3, 256)
    // d_in[2] = node_frame: unused by the reference computation.
    const float* WE   = (const float*)d_in[3];   // (256, 512)
    const float* W1   = (const float*)d_in[4];   // (512, 256)
    const float* b1   = (const float*)d_in[5];   // (256)
    const float* W2   = (const float*)d_in[6];   // (256, 768)
    const float* b2   = (const float*)d_in[7];   // (768)

    const int N = in_sizes[0] / HDIM;            // 100000 (divisible by 16)

    bf16_t* WtE = (bf16_t*)d_ws;                 // 512*256
    bf16_t* Wt1 = WtE + 512 * 256;               // 256*512
    bf16_t* Wt2 = Wt1 + 256 * 512;               // 768*256

    // Transpose + convert weights once (tiny: <0.5M elements).
    const int prep_elems = 512 * 256 + 256 * 512 + 768 * 256;
    prep_weights<<<(prep_elems + 255) / 256, 256, 0, stream>>>(WE, W1, W2, WtE, Wt1, Wt2);

    const size_t smem = (size_t)(3 * 16 * 256 + 16 * 512 + 16 * 256 + 3 * 16 * 256) * sizeof(bf16_t)
                      + (size_t)(16 * 256) * sizeof(float);   // 90112 B (< 320 KB WGP LDS)
    fused_gated_update<<<N / 16, 128, smem, stream>>>(xg, vecg, b1, b2, WtE, Wt1, Wt2,
                                                      (float*)d_out, N);
}
